// TransformerTimewarpLayer_44375602102695
// MI455X (gfx1250) — matrix operand order repaired and checked by reading, not compiled, not run on hardware
//
#include <hip/hip_runtime.h>

// ---------------------------------------------------------------------------
// Problem constants (match reference)
// ---------------------------------------------------------------------------
#define L_SEQ  1024
#define BATCH  4
#define NHEAD  16
#define VDHV   64
#define IN_DIM 1024
#define IN2D   32
#define BHN    (BATCH * NHEAD)   // 64

typedef __attribute__((ext_vector_type(16))) _Float16 v16h;
typedef __attribute__((ext_vector_type(8)))  float    v8f;
typedef unsigned int u32x4 __attribute__((ext_vector_type(4)));
typedef int          i32x4 __attribute__((ext_vector_type(4)));
typedef int          i32x8 __attribute__((ext_vector_type(8)));

union HVec {
  v16h v;
  _Float16 h[16];
  unsigned short s[16];
  uint4 q[2];
};

__device__ __forceinline__ unsigned short f2h(float f) {
  union { _Float16 h; unsigned short u; } c;
  c.h = (_Float16)f;
  return c.u;
}
__device__ __forceinline__ float h2f(unsigned short u) {
  union { unsigned short u; _Float16 h; } c;
  c.u = u;
  return (float)c.h;
}

// ---------------------------------------------------------------------------
// Kernel 1: q/k -> 3-d point coords + squared norm (tiny GEMM, VALU)
// q3[n][l] = {x, y, z, |.|^2}, n = b*H + h
// ---------------------------------------------------------------------------
__global__ __launch_bounds__(256) void k_proj_qk(
    const float* __restrict__ q, const float* __restrict__ k,
    const float* __restrict__ Wq, const float* __restrict__ Wk,
    float4* __restrict__ q3, float4* __restrict__ k3) {
  int idx = blockIdx.x * 256 + threadIdx.x;   // 131072 threads: {q,k} x 4096 rows x 16 heads
  int which = idx >> 16;
  int rem = idx & 0xFFFF;
  int h = rem & 15;
  int lb = rem >> 4;                          // l*BATCH + b
  const float* x = (which ? k : q) + (size_t)lb * IN_DIM;
  const float* W = (which ? Wk : Wq) + (size_t)h * 3 * IN_DIM;
  float a0 = 0.f, a1 = 0.f, a2 = 0.f;
  for (int kk = 0; kk < IN_DIM; kk += 4) {
    float4 xv = *(const float4*)(x + kk);
    float4 w0 = *(const float4*)(W + kk);
    float4 w1 = *(const float4*)(W + IN_DIM + kk);
    float4 w2 = *(const float4*)(W + 2 * IN_DIM + kk);
    a0 += xv.x * w0.x + xv.y * w0.y + xv.z * w0.z + xv.w * w0.w;
    a1 += xv.x * w1.x + xv.y * w1.y + xv.z * w1.z + xv.w * w1.w;
    a2 += xv.x * w2.x + xv.y * w2.y + xv.z * w2.z + xv.w * w2.w;
  }
  int l = lb >> 2, b = lb & 3;
  int n = b * NHEAD + h;
  float4 o = {a0, a1, a2, a0 * a0 + a1 * a1 + a2 * a2};
  (which ? k3 : q3)[(size_t)n * L_SEQ + l] = o;
}

// ---------------------------------------------------------------------------
// Kernel 2: f32 -> f16 conversion of v, Wv, Wout (grid-stride)
// ---------------------------------------------------------------------------
__global__ void k_cvt_f16(const float* __restrict__ v, const float* __restrict__ Wv,
                          const float* __restrict__ Wout,
                          unsigned short* __restrict__ vh,
                          unsigned short* __restrict__ Wvh,
                          unsigned short* __restrict__ Wouth) {
  const int NV = L_SEQ * BATCH * IN_DIM;  // 4M
  const int NW = IN_DIM * IN_DIM;         // 1M
  const int total = NV + 2 * NW;
  for (int i = blockIdx.x * blockDim.x + threadIdx.x; i < total;
       i += gridDim.x * blockDim.x) {
    if (i < NV)            vh[i] = f2h(v[i]);
    else if (i < NV + NW)  Wvh[i - NV] = f2h(Wv[i - NV]);
    else                   Wouth[i - NV - NW] = f2h(Wout[i - NV - NW]);
  }
}

// ---------------------------------------------------------------------------
// Kernel 3: v projection via WMMA, output TRANSPOSED: v3T[n][d][l] (f16)
// tile: M = 16 d-rows (from Wv), N = 16 l-cols (from v), K = 1024
// ---------------------------------------------------------------------------
__global__ __launch_bounds__(256) void k_vproj(
    const unsigned short* __restrict__ vh,
    const unsigned short* __restrict__ Wvh,
    unsigned short* __restrict__ v3T) {
  int wave = threadIdx.x >> 5, lane = threadIdx.x & 31;
  int tile = blockIdx.x * 8 + wave;         // 16384 tiles
  int lt = tile & 63;
  int dt = (tile >> 6) & 3;
  int n  = tile >> 8;
  int bt = n >> 4, h = n & 15;
  int m = lane & 15, hi = lane >> 4;

  const unsigned short* arow = Wvh + (size_t)(h * VDHV + dt * 16 + m) * IN_DIM;
  const unsigned short* brow = vh + ((size_t)(lt * 16 + m) * BATCH + bt) * IN_DIM;

  v8f acc = {};
  for (int kb = 0; kb < IN_DIM; kb += 32) {
    HVec A, B;
    int k0 = kb + hi * 8;                               // A: two 8-elem runs
    A.q[0] = *(const uint4*)(arow + k0);
    A.q[1] = *(const uint4*)(arow + k0 + 16);
    const uint4* bp = (const uint4*)(brow + kb + hi * 8);  // B: two 8-elem runs
    B.q[0] = bp[0]; B.q[1] = bp[2];
    acc = __builtin_amdgcn_wmma_f32_16x16x32_f16(false, A.v, false, B.v,
                                                 (short)0, acc, false, false);
  }
#pragma unroll
  for (int r = 0; r < 8; ++r) {
    int d = dt * 16 + r + hi * 8;    // C-layout: M = r (+8 for hi lanes)
    int l = lt * 16 + m;             // N = lane%16
    v3T[((size_t)n * VDHV + d) * L_SEQ + l] = f2h(acc[r]);
  }
}

// ---------------------------------------------------------------------------
// Kernel 4: pair bias. pbias[n][a][b] = p[bt,a,b,:] . W2d[h] + b2d[h]  (f16)
// One WMMA per 16 (a,b)-rows: M = 16 b-rows, N = 16 heads, K = 32 exactly.
// Pure p-bandwidth kernel (512 MB read).
// ---------------------------------------------------------------------------
__global__ __launch_bounds__(256) void k_pbias(
    const float* __restrict__ p, const float* __restrict__ W2d,
    const float* __restrict__ b2d, unsigned short* __restrict__ pbias) {
  int wave = threadIdx.x >> 5, lane = threadIdx.x & 31;
  int tile = blockIdx.x * 8 + wave;         // 262144 tiles
  int btile = tile & 63;
  int a  = (tile >> 6) & 1023;
  int bt = tile >> 16;
  int m = lane & 15, hi = lane >> 4;

  const float* prow = p + (((size_t)bt * L_SEQ + a) * L_SEQ + btile * 16 + m) * IN2D;
  __builtin_prefetch(prow + 16 * IN2D, 0, 1);   // global_prefetch_b8, next tile

  HVec A, B;
  int k0 = hi * 8;
#pragma unroll
  for (int t = 0; t < 8; ++t) A.s[t] = f2h(prow[k0 + t]);
#pragma unroll
  for (int t = 0; t < 8; ++t) A.s[8 + t] = f2h(prow[k0 + 16 + t]);

  const float* wrow = W2d + (size_t)m * IN2D + hi * 8;   // B[k][h]=W2d[h][k]
#pragma unroll
  for (int t = 0; t < 16; ++t) B.s[t] = f2h(wrow[t + (t & 8)]);

  float cb = b2d[m];                        // bias per output column (head)
  v8f acc;
#pragma unroll
  for (int r = 0; r < 8; ++r) acc[r] = cb;
  acc = __builtin_amdgcn_wmma_f32_16x16x32_f16(false, A.v, false, B.v,
                                               (short)0, acc, false, false);
#pragma unroll
  for (int r = 0; r < 8; ++r) {
    int bcol = btile * 16 + r + hi * 8;     // row M -> b index
    int h = m;                              // col N -> head
    pbias[(((size_t)(bt * NHEAD + h)) * L_SEQ + a) * L_SEQ + bcol] = f2h(acc[r]);
  }
}

// ---------------------------------------------------------------------------
// Kernel 5: fused distance + bias + softmax + attn@V for one (n, 16-row tile)
//  - pair-bias slab (16 x 1024 f16) pulled into LDS by the Tensor Data Mover
//    (one 2D tensor_load_to_lds per WG, TDM LDS-padding of 16B per 1KB gives
//     row stride 1040 f16 -> aligned, low-conflict ds_read_b128 A fetches)
//  - buffer rewritten in place: pb -> logits -> unnormalized exp(P)
//  - P(16x1024)@V(1024x64) via 8 waves: 4 N-tiles x 2 K-halves, LDS reduce
//  - final row rescale by 1/rowsum, store s3 in (L, B, H*VDH) f16 layout
// ---------------------------------------------------------------------------
#define PROW 1040   // 1024 + 2x8 TDM pad shorts (16B per 1024B stored)

__device__ __forceinline__ int pidx(int a, int j) {
  return a * PROW + j + ((j >> 9) << 3);   // mid-row pad of 8 shorts at j=512
}

__global__ __launch_bounds__(256) void k_attn(
    const float4* __restrict__ q3, const float4* __restrict__ k3,
    const unsigned short* __restrict__ pbias,
    const unsigned short* __restrict__ v3T,
    const float* __restrict__ elle,
    unsigned short* __restrict__ s3) {
  __shared__ unsigned short P[16 * PROW];    // 33,280 B
  __shared__ float red[256];
  __shared__ float rowmax[16];
  __shared__ float rowsum[16];
  __shared__ float accx[1024];

  int aTile = blockIdx.x;
  int n = blockIdx.y;
  int bt = n >> 4, h = n & 15;
  int tid = threadIdx.x;
  int t16 = tid & 15, a = tid >> 4;     // 16 threads per logit row
  int aGlob = aTile * 16 + a;

  const unsigned short* pb_tile =
      pbias + ((size_t)n * L_SEQ + (size_t)aTile * 16) * L_SEQ;

  // ---- stage pair-bias tile into LDS -------------------------------------
#if __has_builtin(__builtin_amdgcn_tensor_load_to_lds) && \
    __has_builtin(__builtin_amdgcn_s_wait_tensorcnt)
  if (tid < 32) {   // one TDM op per workgroup (wave 0; EXEC-independent DMA)
    unsigned lds_base = (unsigned)(size_t)(void*)&P[0];
    unsigned long long ga = (unsigned long long)(size_t)pb_tile;
    // D# group 0: count=1, lds_addr, 57-bit global addr, type=2
    u32x4 g0;
    g0[0] = 1u;                                     // count = 1 descriptor
    g0[1] = lds_base;                               // lds_addr
    g0[2] = (unsigned)(ga & 0xFFFFFFFFull);         // global_addr[31:0]
    g0[3] = (unsigned)((ga >> 32) & 0x01FFFFFFull)  // global_addr[56:32]
            | (2u << 30);                           // type = 2 ("image")
    // D# group 1: data_size=1(2B), pad_enable, pad_interval=7 (256 dw = 1KB),
    //             pad_amount=3 (4 dw = 16B); dims: 1024 x (>=16); tile 1024x16
    i32x8 g1;
    g1[0] = (int)((1u << 16) | (1u << 20) | (7u << 22) | (3u << 25));
    g1[1] = (int)((1024u & 0xFFFFu) << 16);         // tensor_dim0[15:0] @ [63:48]
    g1[2] = (int)((1024u & 0xFFFFu) << 16);         // dim0 hi=0 | tensor_dim1 lo
    g1[3] = (int)(1024u << 16);                     // dim1 hi=0 | tile_dim0=1024
    g1[4] = 16;                                     // tile_dim1=16, tile_dim2=0
    g1[5] = 1024;                                   // tensor_dim0_stride lo
    g1[6] = 0;                                      // stride hi | dim1_stride lo
    g1[7] = 0;
    i32x4 gz = {0, 0, 0, 0};                        // 2D: groups 2/3 disabled
#if defined(__clang_major__) && __clang_major__ >= 23
    i32x8 gz8 = {0, 0, 0, 0, 0, 0, 0, 0};
    __builtin_amdgcn_tensor_load_to_lds(g0, g1, gz, gz, gz8, 0);
#else
    __builtin_amdgcn_tensor_load_to_lds(g0, g1, gz, gz, 0);
#endif
    __builtin_amdgcn_s_wait_tensorcnt(0);
  }
#else
  // fallback: cooperative copy global -> LDS (same padded layout)
  for (int j = t16; j < L_SEQ; j += 16)
    P[pidx(a, j)] = pb_tile[(size_t)a * L_SEQ + j];
#endif
  __syncthreads();

  float sp = log1pf(__expf(elle[h]));   // softplus
  float inv_l2 = 1.0f / (sp * sp);

  float4 kv = k3[(size_t)n * L_SEQ + aGlob];
  const float4* qrow = q3 + (size_t)n * L_SEQ;

  // ---- pass 1: logits (in place over pb) + running max -------------------
  float lmax = -3.0e38f;
  for (int j = t16; j < L_SEQ; j += 16) {
    float4 qv = qrow[j];
    float dot = kv.x * qv.x + kv.y * qv.y + kv.z * qv.z;
    float sd = (kv.w + qv.w - 2.0f * dot) * inv_l2;
    int ix = pidx(a, j);
    float logit = h2f(P[ix]) - sd;
    P[ix] = f2h(logit);
    lmax = fmaxf(lmax, logit);
  }
  red[tid] = lmax;
  __syncthreads();
  if (t16 == 0) {
    float mx = red[a * 16];
    for (int t = 1; t < 16; ++t) mx = fmaxf(mx, red[a * 16 + t]);
    rowmax[a] = mx;
  }
  __syncthreads();

  // ---- pass 2: exp + sum (P becomes unnormalized probabilities) ----------
  float mx = rowmax[a];
  float psum = 0.f;
  for (int j = t16; j < L_SEQ; j += 16) {
    int ix = pidx(a, j);
    float e = __expf(h2f(P[ix]) - mx);
    P[ix] = f2h(e);
    psum += e;
  }
  red[tid] = psum;
  __syncthreads();
  if (t16 == 0) {
    float s = 0.f;
    for (int t = 0; t < 16; ++t) s += red[a * 16 + t];
    rowsum[a] = s;
  }
  __syncthreads();

  // ---- pass 3: P @ V via WMMA --------------------------------------------
  int wave = tid >> 5, lane = tid & 31;
  int nt = wave & 3;                     // 16-wide slice of VDH
  int kh = wave >> 2;                    // K half (512 each)
  int mm = lane & 15, hi = lane >> 4;

  const unsigned short* brow = v3T + ((size_t)n * VDHV + nt * 16 + mm) * L_SEQ;
  v8f acc = {};
  for (int kb = kh * 512; kb < kh * 512 + 512; kb += 32) {
    HVec A, B;
    int k0 = kb + hi * 8;                // 8-elem runs never cross the j=512 pad
    A.q[0] = *(const uint4*)&P[pidx(mm, k0)];
    A.q[1] = *(const uint4*)&P[pidx(mm, k0 + 16)];
    const uint4* bp = (const uint4*)(brow + kb + hi * 8);
    B.q[0] = bp[0]; B.q[1] = bp[2];
    acc = __builtin_amdgcn_wmma_f32_16x16x32_f16(false, A.v, false, B.v,
                                                 (short)0, acc, false, false);
  }
  if (kh == 1) {
#pragma unroll
    for (int r = 0; r < 8; ++r) accx[(nt * 8 + r) * 32 + lane] = acc[r];
  }
  __syncthreads();
  if (kh == 0) {
#pragma unroll
    for (int r = 0; r < 8; ++r) {
      acc[r] += accx[(nt * 8 + r) * 32 + lane];
      int row = r + hi * 8;
      float o = acc[r] / rowsum[row];
      int d = nt * 16 + mm;
      // s3 laid out as (L, B, H*VDH) so the final GEMM is a clean row-major A
      s3[((size_t)(aTile * 16 + row) * BATCH + bt) * IN_DIM + h * VDHV + d] =
          f2h(o);
    }
  }
}

// ---------------------------------------------------------------------------
// Kernel 6: out = s3 @ Wout^T + bout via WMMA (4096x1024 @ 1024x1024, f32 out)
// ---------------------------------------------------------------------------
__global__ __launch_bounds__(256) void k_outproj(
    const unsigned short* __restrict__ s3,
    const unsigned short* __restrict__ Wouth,
    const float* __restrict__ bout,
    float* __restrict__ out) {
  int wave = threadIdx.x >> 5, lane = threadIdx.x & 31;
  int tile = blockIdx.x * 8 + wave;     // 16384 tiles
  int jt = tile & 63;
  int rt = tile >> 6;
  int m = lane & 15, hi = lane >> 4;

  const unsigned short* arow = s3 + (size_t)(rt * 16 + m) * IN_DIM;
  const unsigned short* brow = Wouth + (size_t)(jt * 16 + m) * IN_DIM;  // B[k][j]=Wout[j][k]

  float cb = bout[jt * 16 + m];
  v8f acc;
#pragma unroll
  for (int r = 0; r < 8; ++r) acc[r] = cb;

  for (int kb = 0; kb < IN_DIM; kb += 32) {
    HVec A, B;
    int k0 = kb + hi * 8;
    A.q[0] = *(const uint4*)(arow + k0);
    A.q[1] = *(const uint4*)(arow + k0 + 16);
    const uint4* bp = (const uint4*)(brow + kb + hi * 8);
    B.q[0] = bp[0]; B.q[1] = bp[2];
    acc = __builtin_amdgcn_wmma_f32_16x16x32_f16(false, A.v, false, B.v,
                                                 (short)0, acc, false, false);
  }
#pragma unroll
  for (int r = 0; r < 8; ++r) {
    int row = rt * 16 + r + hi * 8;
    out[(size_t)row * IN_DIM + jt * 16 + m] = acc[r];
  }
}

// ---------------------------------------------------------------------------
// Launch
// ---------------------------------------------------------------------------
extern "C" void kernel_launch(void* const* d_in, const int* in_sizes, int n_in,
                              void* d_out, int out_size, void* d_ws, size_t ws_size,
                              hipStream_t stream) {
  const float* q    = (const float*)d_in[0];
  const float* k    = (const float*)d_in[1];
  const float* v    = (const float*)d_in[2];
  const float* p    = (const float*)d_in[3];
  const float* Wq   = (const float*)d_in[4];
  const float* Wk   = (const float*)d_in[5];
  const float* Wv   = (const float*)d_in[6];
  const float* elle = (const float*)d_in[7];
  const float* W2d  = (const float*)d_in[8];
  const float* b2d  = (const float*)d_in[9];
  const float* Wout = (const float*)d_in[10];
  const float* bout = (const float*)d_in[11];
  float* out = (float*)d_out;

  // Workspace layout (~158 MB total)
  char* ws = (char*)d_ws;
  float4* q3            = (float4*)(ws);                                  //   1 MB
  float4* k3            = (float4*)(ws + ((size_t)1 << 20));              //   1 MB
  unsigned short* v3T   = (unsigned short*)(ws + ((size_t)2 << 20));      //   8 MB
  unsigned short* pbias = (unsigned short*)(ws + ((size_t)10 << 20));     // 128 MB
  unsigned short* s3    = (unsigned short*)(ws + ((size_t)138 << 20));    //   8 MB
  unsigned short* vh    = (unsigned short*)(ws + ((size_t)146 << 20));    //   8 MB
  unsigned short* Wvh   = (unsigned short*)(ws + ((size_t)154 << 20));    //   2 MB
  unsigned short* Wouth = (unsigned short*)(ws + ((size_t)156 << 20));    //   2 MB

  k_proj_qk<<<512, 256, 0, stream>>>(q, k, Wq, Wk, q3, k3);
  k_cvt_f16<<<4096, 256, 0, stream>>>(v, Wv, Wout, vh, Wvh, Wouth);
  k_vproj<<<2048, 256, 0, stream>>>(vh, Wvh, v3T);
  k_pbias<<<32768, 256, 0, stream>>>(p, W2d, b2d, pbias);
  k_attn<<<dim3(64, 64), 256, 0, stream>>>(q3, k3, pbias, v3T, elle, s3);
  k_outproj<<<2048, 256, 0, stream>>>(s3, Wouth, bout, out);
}
